// AttnBlock6_b_16784732193436
// MI455X (gfx1250) — matrix operand order repaired and checked by reading, not compiled, not run on hardware
//
#include <hip/hip_runtime.h>
#include <hip/hip_bf16.h>

// ---------------------------------------------------------------------------
// MI455X (gfx1250) implementation of AttnBlock6.
// wave32, WMMA f16 (f32 accum) for all GEMMs, f16 intermediates to halve HBM
// traffic, flash-style global attention (never materializes the 3136x3136
// score matrix), TDM (tensor_load_to_lds) for channel-strided tile staging.
// ---------------------------------------------------------------------------

typedef __attribute__((ext_vector_type(16))) _Float16 v16h;
typedef __attribute__((ext_vector_type(8)))  float    v8f;
typedef __attribute__((ext_vector_type(4)))  unsigned int v4u;
typedef __attribute__((ext_vector_type(8)))  int      v8i;
typedef __attribute__((ext_vector_type(4)))  int      v4i;

constexpr int kC   = 64;
constexpr int kHW  = 200704;   // 448*448
constexpr int kW448 = 448;
constexpr int kP2  = 196;
constexpr int kNG  = 3136;     // 56*56
constexpr int kTG  = 56;
constexpr int kB   = 2;

__device__ __forceinline__ int frag_ka(int j, int hf) {
  // 16-bit A-matrix 16x32 K index for half-pair 2j (ISA 7.12.2 table)
  return (j >> 2) * 16 + (j & 3) * 2 + hf * 8;
}
__device__ __forceinline__ int frag_kb(int j, int hf) {
  // 16-bit B-matrix 32x16 K index, same order as A (ISA 7.12.2 table)
  return (j >> 2) * 16 + (j & 3) * 2 + hf * 8;
}

__device__ __forceinline__ v8f wmma32(v16h a, v16h b, v8f c) {
  return __builtin_amdgcn_wmma_f32_16x16x32_f16(false, a, false, b,
                                                (short)0, c, false, false);
}

// ---------------------------------------------------------------------------
// Tensor Data Mover: DMA a 2D tile (tile_w contiguous elems x tile_h rows,
// row stride = row_stride elems) from global memory into LDS.
// Descriptor bitfields per CDNA5 ISA.
// data_size_code: 0=1B 1=2B 2=4B 3=8B.  Issue from ONE wave per WG (EXEC is
// ignored by TDM), s_wait_tensorcnt 0, then barrier before consumers read.
// ---------------------------------------------------------------------------
__device__ __forceinline__ void tdm_load_2d_to_lds(
    unsigned lds_byte_off, const void* gva, unsigned tile_w, unsigned tile_h,
    unsigned row_stride, unsigned data_size_code) {
  unsigned long long ga = (unsigned long long)(uintptr_t)gva;
  v4u g0;
  g0[0] = 1u;                                               // count=1 (valid)
  g0[1] = lds_byte_off;                                     // lds_addr
  g0[2] = (unsigned)ga;                                     // global_addr lo
  g0[3] = (unsigned)((ga >> 32) & 0x01ffffffu) | (2u << 30);// addr hi | type=2
  v8i g1;
  g1[0] = (int)(data_size_code << 16);     // wg_mask=0 | data_size
  g1[1] = (int)(tile_w << 16);             // tensor_dim0[15:0]
  g1[2] = (int)(tile_h << 16);             // dim0 hi=0 | tensor_dim1[15:0]
  g1[3] = (int)(tile_w << 16);             // dim1 hi=0 | tile_dim0
  g1[4] = (int)(tile_h & 0xffffu);         // tile_dim1 | tile_dim2=0
  g1[5] = (int)row_stride;                 // tensor_dim0_stride lo
  g1[6] = 0;                               // stride hi | dim1_stride lo
  g1[7] = 0;
  v4i gz = {0, 0, 0, 0};
#if __has_include(<hip/amd_detail/amd_gfx1250_TDM.h>)
  v8i gz8 = {0, 0, 0, 0, 0, 0, 0, 0};
  __builtin_amdgcn_tensor_load_to_lds(g0, g1, gz, gz, gz8, 0);
#else
  __builtin_amdgcn_tensor_load_to_lds(g0, g1, gz, gz, 0);
#endif
}

// ---------------------------------------------------------------------------
// Workspace layout
// ---------------------------------------------------------------------------
constexpr size_t align_up(size_t x) { return (x + 255) & ~(size_t)255; }
constexpr size_t OFF_ACC  = 0;                                  // 4 f32
constexpr size_t OFF_STAT = 256;                                // 4 f32
constexpr size_t SZ_QKV   = align_up((size_t)kB * kC * kHW * 2 + 256); // f16 (+OOB pad)
constexpr size_t OFF_QH   = 512;
constexpr size_t OFF_KH   = OFF_QH + SZ_QKV;
constexpr size_t OFF_VH   = OFF_KH + SZ_QKV;
constexpr size_t SZ_G     = align_up((size_t)kB * kC * kNG * 2);
constexpr size_t OFF_QG   = OFF_VH + SZ_QKV;
constexpr size_t OFF_KG   = OFF_QG + SZ_G;
constexpr size_t OFF_VG   = OFF_KG + SZ_G;
constexpr size_t OFF_HG   = OFF_VG + SZ_G;                      // f32
constexpr size_t OFF_WM   = OFF_HG + align_up((size_t)kB * kC * kNG * 4);
constexpr size_t OFF_WMH  = OFF_WM + align_up((size_t)kB * 208 * 208 * 4);
constexpr size_t OFF_HB   = OFF_WMH + align_up((size_t)kB * 208 * 224 * 2);

// ---------------------------------------------------------------------------
// 0. zero the stat accumulators (every call; deterministic)
// ---------------------------------------------------------------------------
__global__ void k_init(float* acc) {
  if (threadIdx.x < 4) acc[threadIdx.x] = 0.f;
}

// ---------------------------------------------------------------------------
// 1. GroupNorm statistics: per-batch sum / sumsq
// ---------------------------------------------------------------------------
__global__ __launch_bounds__(256) void k_stats(const float* __restrict__ x,
                                               float* __restrict__ acc) {
  int b = blockIdx.y;
  const float* xb = x + (size_t)b * kC * kHW;
  size_t N = (size_t)kC * kHW;
  float s = 0.f, s2 = 0.f;
  for (size_t i = (size_t)blockIdx.x * blockDim.x + threadIdx.x; i < N;
       i += (size_t)gridDim.x * blockDim.x) {
    float v = xb[i];
    s += v; s2 += v * v;
  }
  __shared__ float ls[256], ls2[256];
  ls[threadIdx.x] = s; ls2[threadIdx.x] = s2;
  __syncthreads();
  for (int off = 128; off > 0; off >>= 1) {
    if ((int)threadIdx.x < off) {
      ls[threadIdx.x]  += ls[threadIdx.x + off];
      ls2[threadIdx.x] += ls2[threadIdx.x + off];
    }
    __syncthreads();
  }
  if (threadIdx.x == 0) {
    atomicAdd(&acc[b * 2 + 0], ls[0]);
    atomicAdd(&acc[b * 2 + 1], ls2[0]);
  }
}

__global__ void k_finalize(const float* __restrict__ acc, float* __restrict__ stats) {
  int b = threadIdx.x;
  if (b < kB) {
    float N = (float)kC * (float)kHW;
    float mean = acc[b * 2 + 0] / N;
    float var  = acc[b * 2 + 1] / N - mean * mean;
    stats[b * 2 + 0] = mean;
    stats[b * 2 + 1] = rsqrtf(var + 1e-5f);
  }
}

// ---------------------------------------------------------------------------
// 2. Fused GroupNorm + QKV 1x1 conv (WMMA).  WG: 8 waves x 16 pixels.
//    x tile (64ch x 128px, stride HW) staged through TDM into LDS.
// ---------------------------------------------------------------------------
__global__ __launch_bounds__(256) void k_qkv(
    const float* __restrict__ x, const float* __restrict__ gnw,
    const float* __restrict__ gnb,
    const float* __restrict__ qw, const float* __restrict__ qbi,
    const float* __restrict__ kw, const float* __restrict__ kbi,
    const float* __restrict__ vw, const float* __restrict__ vbi,
    const float* __restrict__ stats,
    _Float16* __restrict__ qh, _Float16* __restrict__ kh,
    _Float16* __restrict__ vh) {
  __shared__ _Float16 W[3][64][64];
  __shared__ float bias[3][64];
  __shared__ float lX[64][128];

  int t = threadIdx.x, lane = t & 31, w = t >> 5;
  int nn = lane & 15, hf = lane >> 4;
  int b = blockIdx.y;
  int pixwg = blockIdx.x * 128;
  const float* xb = x + (size_t)b * kC * kHW;

  if (w == 0)
    tdm_load_2d_to_lds((unsigned)(uintptr_t)&lX[0][0], xb + pixwg,
                       128u, 64u, (unsigned)kHW, 2u);

  for (int i = t; i < 4096; i += 256) {
    W[0][i >> 6][i & 63] = (_Float16)qw[i];
    W[1][i >> 6][i & 63] = (_Float16)kw[i];
    W[2][i >> 6][i & 63] = (_Float16)vw[i];
  }
  if (t < 64) { bias[0][t] = qbi[t]; bias[1][t] = kbi[t]; bias[2][t] = vbi[t]; }
  if (w == 0) __builtin_amdgcn_s_wait_tensorcnt(0);
  __syncthreads();

  int col = w * 16 + nn;
  int pix = pixwg + col;
  float mean = stats[b * 2 + 0], rstd = stats[b * 2 + 1];

  v16h bf[2];
  for (int ch = 0; ch < 2; ++ch)
    for (int j = 0; j < 8; ++j) {
      int c0 = ch * 32 + frag_kb(j, hf);
      float v0 = (lX[c0][col] - mean) * rstd * gnw[c0] + gnb[c0];
      float v1 = (lX[c0 + 1][col] - mean) * rstd * gnw[c0 + 1] + gnb[c0 + 1];
      bf[ch][2 * j]     = (_Float16)v0;
      bf[ch][2 * j + 1] = (_Float16)v1;
    }

  for (int mat = 0; mat < 3; ++mat) {
    _Float16* dst = (mat == 0 ? qh : mat == 1 ? kh : vh) + (size_t)b * kC * kHW;
    for (int mt = 0; mt < 4; ++mt) {
      v16h a0, a1;
      int m = mt * 16 + nn;
      for (int j = 0; j < 8; ++j) {
        int k = frag_ka(j, hf);
        a0[2 * j] = W[mat][m][k];      a0[2 * j + 1] = W[mat][m][k + 1];
        a1[2 * j] = W[mat][m][32 + k]; a1[2 * j + 1] = W[mat][m][32 + k + 1];
      }
      v8f acc = {};
      acc = wmma32(a0, bf[0], acc);
      acc = wmma32(a1, bf[1], acc);
      for (int j = 0; j < 8; ++j) {
        int o = mt * 16 + j + 8 * hf;
        dst[(size_t)o * kHW + pix] = (_Float16)(acc[j] + bias[mat][o]);
      }
    }
  }
}

// ---------------------------------------------------------------------------
// 3. 8x8 average pool of q,k,v -> qg,kg,vg [b,64,3136] (f16)
// ---------------------------------------------------------------------------
__global__ __launch_bounds__(256) void k_pool(
    const _Float16* __restrict__ qh, const _Float16* __restrict__ kh,
    const _Float16* __restrict__ vh,
    _Float16* __restrict__ qg, _Float16* __restrict__ kg,
    _Float16* __restrict__ vg) {
  int idx = blockIdx.x * blockDim.x + threadIdx.x;
  if (idx >= kB * kC * kNG) return;
  int g = idx % kNG, c = (idx / kNG) % kC, b = idx / (kNG * kC);
  int ty = g / kTG, tx = g % kTG;
  size_t base = ((size_t)b * kC + c) * kHW + (size_t)ty * 8 * kW448 + (size_t)tx * 8;
  float sq = 0.f, sk = 0.f, sv = 0.f;
  for (int dy = 0; dy < 8; ++dy)
    for (int dx = 0; dx < 8; ++dx) {
      size_t a = base + (size_t)dy * kW448 + dx;
      sq += (float)qh[a]; sk += (float)kh[a]; sv += (float)vh[a];
    }
  qg[idx] = (_Float16)(sq * (1.f / 64.f));
  kg[idx] = (_Float16)(sk * (1.f / 64.f));
  vg[idx] = (_Float16)(sv * (1.f / 64.f));
}

// ---------------------------------------------------------------------------
// 4. Patch-attention logits: wm[p,q] = sum_{s=0..65535} q[s,p]*k[s,q] / 256
//    WG = one 16x16 output tile; 8 waves split-K; LDS-transposed f16 staging.
// ---------------------------------------------------------------------------
__global__ __launch_bounds__(256) void k_patch_logits(
    const _Float16* __restrict__ qh, const _Float16* __restrict__ kh,
    float* __restrict__ wm) {
  int tile = blockIdx.x;               // 0..168 (13x13)
  int pt = tile / 13, qt = tile % 13;
  int b = blockIdx.y;
  int t = threadIdx.x, lane = t & 31, w = t >> 5;
  int nn = lane & 15, hf = lane >> 4;

  __shared__ _Float16 lA[8][16][32];   // [wave][p][s_local]
  __shared__ _Float16 lB[8][16][32];   // [wave][q][s_local]
  __shared__ float red[8][256];

  const _Float16* qb_ = qh + (size_t)b * kC * kHW;
  const _Float16* kb_ = kh + (size_t)b * kC * kHW;
  v8f acc = {};

  for (int it = 0; it < 256; ++it) {
    int s = w * 8192 + it * 32 + lane;
    int c = s >> 10, chk = s & 1023;
    size_t rowq = (size_t)c * kHW + (size_t)chk * kP2 + pt * 16;
    size_t rowk = (size_t)c * kHW + (size_t)chk * kP2 + qt * 16;
    if (it + 1 < 256) {                // software prefetch next K slice
      int s2 = s + 32;
      int c2 = s2 >> 10, chk2 = s2 & 1023;
      __builtin_prefetch(qb_ + (size_t)c2 * kHW + (size_t)chk2 * kP2 + pt * 16, 0, 3);
      __builtin_prefetch(kb_ + (size_t)c2 * kHW + (size_t)chk2 * kP2 + qt * 16, 0, 3);
    }
    __syncthreads();                   // prior fragment reads complete
    for (int p = 0; p < 16; ++p) {
      lA[w][p][lane] = qb_[rowq + p];
      lB[w][p][lane] = kb_[rowk + p];
    }
    __syncthreads();
    v16h a, bb;
    const _Float16* ra = &lA[w][nn][0];
    const _Float16* rb = &lB[w][nn][0];
    for (int j = 0; j < 8; ++j) {
      int ka = frag_ka(j, hf);
      int kb2 = frag_kb(j, hf);
      a[2 * j] = ra[ka];   a[2 * j + 1] = ra[ka + 1];
      bb[2 * j] = rb[kb2]; bb[2 * j + 1] = rb[kb2 + 1];
    }
    acc = wmma32(a, bb, acc);
  }

  __syncthreads();
  for (int j = 0; j < 8; ++j) red[w][lane * 8 + j] = acc[j];
  __syncthreads();
  if (w == 0) {
    for (int tt = lane; tt < 256; tt += 32) {
      int l = tt / 8, j = tt % 8;
      float sum = 0.f;
      for (int ww = 0; ww < 8; ++ww) sum += red[ww][tt];
      int m = j + 8 * (l >> 4), n2 = l & 15;
      int p = pt * 16 + m, q = qt * 16 + n2;
      if (p < kP2 && q < kP2)
        wm[((size_t)b * 208 + p) * 208 + q] = sum * (1.f / 256.f);
    }
  }
}

// ---------------------------------------------------------------------------
// 5. Row softmax (196 wide) -> padded f16 prob matrix [b][208][224] (pads = 0)
// ---------------------------------------------------------------------------
__global__ __launch_bounds__(256) void k_softmax(const float* __restrict__ wm,
                                                 _Float16* __restrict__ wmh) {
  int row = blockIdx.x, b = blockIdx.y;
  _Float16* out = wmh + ((size_t)b * 208 + row) * 224;
  if (row >= kP2) {
    for (int k = threadIdx.x; k < 224; k += 256) out[k] = (_Float16)0.f;
    return;
  }
  const float* in = wm + ((size_t)b * 208 + row) * 208;
  int t = threadIdx.x;
  float v = (t < kP2) ? in[t] : -1e30f;
  __shared__ float sm[256];
  sm[t] = v; __syncthreads();
  for (int off = 128; off > 0; off >>= 1) {
    if (t < off) sm[t] = fmaxf(sm[t], sm[t + off]);
    __syncthreads();
  }
  float mx = sm[0]; __syncthreads();
  float e = (t < kP2) ? __expf(v - mx) : 0.f;
  sm[t] = e; __syncthreads();
  for (int off = 128; off > 0; off >>= 1) {
    if (t < off) sm[t] += sm[t + off];
    __syncthreads();
  }
  float inv = 1.f / sm[0];
  if (t < 224) out[t] = (_Float16)((t < kP2) ? e * inv : 0.f);
}

// ---------------------------------------------------------------------------
// 6. Global pooled attention, flash-style.  4 waves/WG, one 16-query tile per
//    wave, online softmax over 196 key tiles, PV via WMMA (P transposed
//    through per-wave LDS into A-fragment layout).
// ---------------------------------------------------------------------------
__global__ __launch_bounds__(128) void k_gattn(
    const _Float16* __restrict__ qg, const _Float16* __restrict__ kg,
    const _Float16* __restrict__ vg, float* __restrict__ hg) {
  int b = blockIdx.y;
  int t = threadIdx.x, lane = t & 31, w = t >> 5;
  int nn = lane & 15, hf = lane >> 4;
  int qbase = (blockIdx.x * 4 + w) * 16;

  const _Float16* qgb = qg + (size_t)b * kC * kNG;
  const _Float16* kgb = kg + (size_t)b * kC * kNG;
  const _Float16* vgb = vg + (size_t)b * kC * kNG;

  v16h aq[2];
  for (int ch = 0; ch < 2; ++ch)
    for (int j = 0; j < 8; ++j) {
      int c = ch * 32 + frag_ka(j, hf);
      aq[ch][2 * j]     = qgb[(size_t)c * kNG + qbase + nn];
      aq[ch][2 * j + 1] = qgb[(size_t)(c + 1) * kNG + qbase + nn];
    }

  float rmax[8], rsum[8];
  v8f zero = {};
  v8f o[4];
  for (int i = 0; i < 4; ++i) o[i] = zero;
  for (int j = 0; j < 8; ++j) { rmax[j] = -1e30f; rsum[j] = 0.f; }

  __shared__ _Float16 lP[4][16][32];

  for (int kt = 0; kt < 196; kt += 2) {
    v8f s[2];
    for (int tt = 0; tt < 2; ++tt) {
      int kbase = (kt + tt) * 16;
      v16h bk[2];
      for (int ch = 0; ch < 2; ++ch)
        for (int j = 0; j < 8; ++j) {
          int c = ch * 32 + frag_kb(j, hf);
          bk[ch][2 * j]     = kgb[(size_t)c * kNG + kbase + nn];
          bk[ch][2 * j + 1] = kgb[(size_t)(c + 1) * kNG + kbase + nn];
        }
      v8f z = zero;
      z = wmma32(aq[0], bk[0], z);
      z = wmma32(aq[1], bk[1], z);
      s[tt] = z;
    }
    __syncthreads();                  // prior lP reads complete
    for (int j = 0; j < 8; ++j) {
      float s0 = s[0][j] * 0.125f, s1 = s[1][j] * 0.125f;
      float tm = fmaxf(s0, s1);
      for (int off = 1; off < 16; off <<= 1) tm = fmaxf(tm, __shfl_xor(tm, off, 32));
      float nm = fmaxf(rmax[j], tm);
      float sc = __expf(rmax[j] - nm);
      float p0 = __expf(s0 - nm), p1 = __expf(s1 - nm);
      float rs = p0 + p1;
      for (int off = 1; off < 16; off <<= 1) rs += __shfl_xor(rs, off, 32);
      rsum[j] = rsum[j] * sc + rs;
      rmax[j] = nm;
      for (int ct = 0; ct < 4; ++ct) o[ct][j] *= sc;
      int m = j + 8 * hf;
      lP[w][m][nn]      = (_Float16)p0;
      lP[w][m][16 + nn] = (_Float16)p1;
    }
    __syncthreads();
    v16h ap;
    {
      const _Float16* rp = &lP[w][nn][0];
      for (int j = 0; j < 8; ++j) {
        int k = frag_ka(j, hf);
        ap[2 * j] = rp[k]; ap[2 * j + 1] = rp[k + 1];
      }
    }
    for (int ct = 0; ct < 4; ++ct) {
      v16h bv;
      int c = ct * 16 + nn;
      const _Float16* vr = vgb + (size_t)c * kNG + kt * 16;
      for (int j = 0; j < 8; ++j) {
        int k = frag_kb(j, hf);
        bv[2 * j] = vr[k]; bv[2 * j + 1] = vr[k + 1];
      }
      o[ct] = wmma32(ap, bv, o[ct]);
    }
  }

  float* hb = hg + (size_t)b * kC * kNG;
  for (int ct = 0; ct < 4; ++ct) {
    int c = ct * 16 + nn;
    for (int j = 0; j < 8; ++j) {
      int q = qbase + j + 8 * hf;
      hb[(size_t)c * kNG + q] = o[ct][j] / rsum[j];
    }
  }
}

// ---------------------------------------------------------------------------
// 7. Patch-attention apply fused with 0.75/0.25 combine + nearest upsample.
//    WG = one 196-pixel chunk; v tile (64x196 f16, stride HW) staged via TDM.
//    hp[q',c] = sum_k wm[q',k] * v[c,k] via WMMA.  K-tail over-reads hit
//    zero columns of wmh, so products vanish (v values are finite).
// ---------------------------------------------------------------------------
__global__ __launch_bounds__(256) void k_patch_apply(
    const _Float16* __restrict__ vh, const _Float16* __restrict__ wmh,
    const float* __restrict__ hg, _Float16* __restrict__ hbuf) {
  int chunk = blockIdx.x;              // 0..1023
  int b = blockIdx.y;
  int t = threadIdx.x, lane = t & 31, w = t >> 5;
  int nn = lane & 15, hf = lane >> 4;

  __shared__ _Float16 lV[64 * 196 + 64];   // v[c][k] row stride 196 (+ slack)
  const _Float16* vb = vh + (size_t)b * kC * kHW + (size_t)chunk * kP2;
  if (w == 0) {
    tdm_load_2d_to_lds((unsigned)(uintptr_t)&lV[0], vb,
                       (unsigned)kP2, 64u, (unsigned)kHW, 1u);
    __builtin_amdgcn_s_wait_tensorcnt(0);
  }
  __syncthreads();

  const _Float16* wrow = wmh + (size_t)b * 208 * 224;
  for (int tile = w; tile < 52; tile += 8) {   // 13 m-tiles x 4 c-tiles
    int mt = tile >> 2, ct = tile & 3;
    v8f acc = {};
    for (int kc = 0; kc < 7; ++kc) {
      v16h a, bb;
      const _Float16* ra = wrow + (size_t)(mt * 16 + nn) * 224 + kc * 32;
      const _Float16* rb = &lV[(ct * 16 + nn) * kP2 + kc * 32];
      for (int j = 0; j < 8; ++j) {
        int ka = frag_ka(j, hf), kb2 = frag_kb(j, hf);
        a[2 * j] = ra[ka];   a[2 * j + 1] = ra[ka + 1];
        bb[2 * j] = rb[kb2]; bb[2 * j + 1] = rb[kb2 + 1];
      }
      acc = wmma32(a, bb, acc);
    }
    int c = ct * 16 + nn;
    const float* hgb = hg + ((size_t)b * kC + c) * kNG;
    _Float16* hb = hbuf + ((size_t)b * kC + c) * kHW + (size_t)chunk * kP2;
    for (int j = 0; j < 8; ++j) {
      int qq = mt * 16 + j + 8 * hf;
      if (qq < kP2) {
        int pixel = chunk * kP2 + qq;
        int y = pixel / kW448, xc = pixel % kW448;
        float gv = hgb[(y >> 3) * kTG + (xc >> 3)];
        hb[qq] = (_Float16)(0.75f * acc[j] + 0.25f * gv);
      }
    }
  }
}

// ---------------------------------------------------------------------------
// 8. proj 1x1 conv + residual (WMMA), out = x + W_proj @ h_in.
//    h tile (64x128 f16, stride HW) staged via TDM.
// ---------------------------------------------------------------------------
__global__ __launch_bounds__(256) void k_proj(
    const _Float16* __restrict__ hbuf, const float* __restrict__ pw,
    const float* __restrict__ x, float* __restrict__ out) {
  __shared__ _Float16 W[64][64];
  __shared__ _Float16 lH[64][128];
  int t = threadIdx.x, lane = t & 31, w = t >> 5;
  int nn = lane & 15, hf = lane >> 4;
  int b = blockIdx.y;
  int pixwg = blockIdx.x * 128;
  const _Float16* hb = hbuf + (size_t)b * kC * kHW;

  if (w == 0)
    tdm_load_2d_to_lds((unsigned)(uintptr_t)&lH[0][0], hb + pixwg,
                       128u, 64u, (unsigned)kHW, 1u);
  for (int i = t; i < 4096; i += 256) W[i >> 6][i & 63] = (_Float16)pw[i];
  if (w == 0) __builtin_amdgcn_s_wait_tensorcnt(0);
  __syncthreads();

  int col = w * 16 + nn;
  int pix = pixwg + col;

  v16h bf[2];
  for (int ch = 0; ch < 2; ++ch)
    for (int j = 0; j < 8; ++j) {
      int c = ch * 32 + frag_kb(j, hf);
      bf[ch][2 * j]     = lH[c][col];
      bf[ch][2 * j + 1] = lH[c + 1][col];
    }

  const float* xb = x + (size_t)b * kC * kHW;
  float* ob = out + (size_t)b * kC * kHW;
  for (int mt = 0; mt < 4; ++mt) {
    v16h a0, a1;
    int m = mt * 16 + nn;
    for (int j = 0; j < 8; ++j) {
      int k = frag_ka(j, hf);
      a0[2 * j] = W[m][k];      a0[2 * j + 1] = W[m][k + 1];
      a1[2 * j] = W[m][32 + k]; a1[2 * j + 1] = W[m][32 + k + 1];
    }
    v8f acc = {};
    acc = wmma32(a0, bf[0], acc);
    acc = wmma32(a1, bf[1], acc);
    for (int j = 0; j < 8; ++j) {
      int o = mt * 16 + j + 8 * hf;
      size_t idx = (size_t)o * kHW + pix;
      ob[idx] = xb[idx] + acc[j];
    }
  }
}

// ---------------------------------------------------------------------------
extern "C" void kernel_launch(void* const* d_in, const int* in_sizes, int n_in,
                              void* d_out, int out_size, void* d_ws, size_t ws_size,
                              hipStream_t stream) {
  const float* x   = (const float*)d_in[0];
  const float* gnw = (const float*)d_in[1];
  const float* gnb = (const float*)d_in[2];
  const float* qw  = (const float*)d_in[3];
  const float* qb  = (const float*)d_in[4];
  const float* kw  = (const float*)d_in[5];
  const float* kb  = (const float*)d_in[6];
  const float* vw  = (const float*)d_in[7];
  const float* vb  = (const float*)d_in[8];
  const float* pw  = (const float*)d_in[9];
  float* out = (float*)d_out;

  char* ws = (char*)d_ws;
  float*    acc   = (float*)(ws + OFF_ACC);
  float*    stats = (float*)(ws + OFF_STAT);
  _Float16* qh    = (_Float16*)(ws + OFF_QH);
  _Float16* kh    = (_Float16*)(ws + OFF_KH);
  _Float16* vh    = (_Float16*)(ws + OFF_VH);
  _Float16* qg    = (_Float16*)(ws + OFF_QG);
  _Float16* kg    = (_Float16*)(ws + OFF_KG);
  _Float16* vg    = (_Float16*)(ws + OFF_VG);
  float*    hg    = (float*)(ws + OFF_HG);
  float*    wm    = (float*)(ws + OFF_WM);
  _Float16* wmh   = (_Float16*)(ws + OFF_WMH);
  _Float16* hbuf  = (_Float16*)(ws + OFF_HB);

  k_init<<<1, 64, 0, stream>>>(acc);
  k_stats<<<dim3(512, kB), 256, 0, stream>>>(x, acc);
  k_finalize<<<1, 64, 0, stream>>>(acc, stats);
  k_qkv<<<dim3(1568, kB), 256, 0, stream>>>(x, gnw, gnb, qw, qb, kw, kb, vw, vb,
                                            stats, qh, kh, vh);
  k_pool<<<dim3(1568), 256, 0, stream>>>(qh, kh, vh, qg, kg, vg);
  k_patch_logits<<<dim3(169, kB), 256, 0, stream>>>(qh, kh, wm);
  k_softmax<<<dim3(208, kB), 256, 0, stream>>>(wm, wmh);
  k_gattn<<<dim3(49, kB), 128, 0, stream>>>(qg, kg, vg, hg);
  k_patch_apply<<<dim3(1024, kB), 256, 0, stream>>>(vh, wmh, hg, hbuf);
  k_proj<<<dim3(1568, kB), 256, 0, stream>>>(hbuf, pw, x, out);
}
